// PairInteractionGrid_16484084483179
// MI455X (gfx1250) — matrix operand order repaired and checked by reading, not compiled, not run on hardware
//
#include <hip/hip_runtime.h>

// ---------------------------------------------------------------------------
// PairInteractionGrid for MI455X (gfx1250, wave32, WMMA f32_16x16x32_f16).
//
// Shapes: B=4 T=64 N=M=32 D=128.  out = (B,T,N*M,D) f32 = 128 MiB.
// Factorization:  pair@W1 = p@W1a (n-only) + v@W1b (m-only)
//                          + (p*v)@W1c + |p-v|@W1d     (pair-dependent)
// Kernel 1: per (b,t): p, v (f16 -> ws) and PA=p@W1a, VB=v@W1b (f32 -> ws),
//           results staged in LDS and streamed out with async b128 stores.
// Kernel 2: per (b,t, pair segment): per-pair WMMA on prod/absdiff + W2,
//           fused SiLU; inputs pulled in with GLOBAL_LOAD_ASYNC_TO_LDS,
//           output tiles staged in LDS and streamed out with
//           GLOBAL_STORE_ASYNC_FROM_LDS_B128 (full-cacheline traffic).
// Workspace use: 12 MiB.
// ---------------------------------------------------------------------------

typedef _Float16 half_t;
typedef __attribute__((ext_vector_type(16))) _Float16 v16h;
typedef __attribute__((ext_vector_type(8)))  _Float16 v8h;
typedef __attribute__((ext_vector_type(8)))  float    v8f;

#define NTHREADS 256

__device__ __forceinline__ v8f wmma16(v16h a, v16h b, v8f c) {
  return __builtin_amdgcn_wmma_f32_16x16x32_f16(
      /*neg_a=*/false, a, /*neg_b=*/false, b,
      /*c_mod=*/(short)0, c, /*reuse_a=*/false, /*reuse_b=*/false);
}

// --- CDNA5 async LDS<->global path (ASYNCcnt) ------------------------------
__device__ __forceinline__ void async_load_b128(uint32_t lds_off, const char* g) {
  asm volatile("global_load_async_to_lds_b128 %0, %1, off"
               :: "v"(lds_off), "v"(g) : "memory");
}
__device__ __forceinline__ void async_store_b128(const char* g, uint32_t lds_off) {
  asm volatile("global_store_async_from_lds_b128 %0, %1, off"
               :: "v"(g), "v"(lds_off) : "memory");
}
__device__ __forceinline__ void wait_async0() {
  asm volatile("s_wait_asynccnt 0" ::: "memory");
}
__device__ __forceinline__ void wait_ds0() {
  asm volatile("s_wait_dscnt 0" ::: "memory");
}
// Cooperative block-wide copies (contiguous, 16B per lane per op).
__device__ __forceinline__ void asyncG2L(uint32_t ldsoff, const void* gsrc,
                                         int nbytes, int tid) {
  const char* g = (const char*)gsrc;
  for (int o = tid * 16; o < nbytes; o += NTHREADS * 16)
    async_load_b128(ldsoff + (uint32_t)o, g + o);
}
__device__ __forceinline__ void asyncL2G(void* gdst, uint32_t ldsoff,
                                         int nbytes, int tid) {
  char* g = (char*)gdst;
  for (int o = tid * 16; o < nbytes; o += NTHREADS * 16)
    async_store_b128(g + o, ldsoff + (uint32_t)o);
}

__device__ __forceinline__ v16h cat8(v8h lo, v8h hi) {
  v16h r;
#pragma unroll
  for (int i = 0; i < 8; ++i) { r[i] = lo[i]; r[i + 8] = hi[i]; }
  return r;
}

// A-fragment (16x32 f16, wave32): lane row = lane&15,
// halves 0..7  -> K = kk*32 + klo + 0..7
// halves 8..15 -> K = kk*32 + klo + 16..23,   klo = (lane<16) ? 0 : 8.
__device__ __forceinline__ v16h loadArow(const half_t* row, int kk, int lane) {
  int d0 = kk * 32 + ((lane & 16) ? 8 : 0);
  v8h lo = *(const v8h*)(row + d0);
  v8h hi = *(const v8h*)(row + d0 + 16);
  return cat8(lo, hi);
}

// prod / absdiff A-fragments built in registers from p-row (shared, fixed n)
// and v-row (per-lane m).
__device__ __forceinline__ void buildPV(const half_t* prow, const half_t* vrow,
                                        int kk, int lane, v16h& aP, v16h& aD) {
  int d0 = kk * 32 + ((lane & 16) ? 8 : 0);
  v8h pl = *(const v8h*)(prow + d0);
  v8h ph = *(const v8h*)(prow + d0 + 16);
  v8h vl = *(const v8h*)(vrow + d0);
  v8h vh = *(const v8h*)(vrow + d0 + 16);
#pragma unroll
  for (int i = 0; i < 8; ++i) {
    half_t p0 = pl[i], p1 = ph[i], q0 = vl[i], q1 = vh[i];
    aP[i]     = p0 * q0;
    aP[i + 8] = p1 * q1;
    half_t a0 = p0 - q0; if (a0 < (half_t)0) a0 = -a0;
    half_t a1 = p1 - q1; if (a1 < (half_t)0) a1 = -a1;
    aD[i]     = a0;
    aD[i + 8] = a1;
  }
}

// B-fragment (32x16 f16): lane col = lane&15,
// halves i -> K = kk*32 + (i&7) + (i<8 ? 0 : 16) + (lane<16 ? 0 : 8).
// Weights pre-swizzled in LDS as [c][kk][lane][16 halves]: 2x ds_load_b128.
__device__ __forceinline__ v16h loadB(const half_t* sw, int c, int kk, int lane) {
  const half_t* p = sw + (((c * 4 + kk) * 32 + lane) << 4);
  v8h lo = *(const v8h*)p;
  v8h hi = *(const v8h*)(p + 8);
  return cat8(lo, hi);
}

// Swizzle a 128x128 f32 weight block (row-major, K x Ncol) into the
// fragment-order f16 LDS image described above. 8192 u32 slots.
__device__ __forceinline__ void swizzleW(const float* __restrict__ Wsrc,
                                         half_t* dst, int tid) {
  uint32_t* d32 = (uint32_t*)dst;
  for (int i = tid; i < 8192; i += NTHREADS) {
    int hp = i & 7;            // half-pair
    int l  = (i >> 3) & 31;    // lane
    int kk = (i >> 8) & 3;
    int c  = (i >> 10);
    int K  = kk * 32 + hp * 2 + ((hp & 4) ? 8 : 0) + ((l & 16) ? 8 : 0);
    int col = c * 16 + (l & 15);
    union { uint32_t u; half_t f[2]; } pk;
    pk.f[0] = (half_t)Wsrc[K * 128 + col];
    pk.f[1] = (half_t)Wsrc[(K + 1) * 128 + col];
    d32[i] = pk.u;
  }
}

__device__ __forceinline__ void cvtCopyF16(const float* __restrict__ src,
                                           half_t* dst, int nElem, int tid) {
  uint32_t* d32 = (uint32_t*)dst;
  for (int i = tid; i < (nElem >> 1); i += NTHREADS) {
    union { uint32_t u; half_t f[2]; } pk;
    pk.f[0] = (half_t)src[2 * i];
    pk.f[1] = (half_t)src[2 * i + 1];
    d32[i] = pk.u;
  }
}

__device__ __forceinline__ void copyF32(const float* __restrict__ src,
                                        float* dst, int n, int tid) {
  for (int i = tid; i < n; i += NTHREADS) dst[i] = src[i];
}

// ---------------------------------------------------------------------------
// Kernel 1 LDS layout (byte offsets from dynamic-LDS base):
//   0       sWp   32KB   (phase 2 reuses [0,32KB)   as PA f32 staging)
//   32768   sWv   32KB   (phase 2 reuses [32K,64K)  as VB f32 staging)
//   65536   sW1a  32KB
//   98304   sW1b  32KB
//   131072  sX    8KB    147456 sP 8KB
//   139264  sY    8KB    155648 sV 8KB  (sP/sV stream out async)
//   163840  sbp/sbv 1KB                  total 164864
// ---------------------------------------------------------------------------
#define K1_WP   0u
#define K1_WV   32768u
#define K1_W1A  65536u
#define K1_W1B  98304u
#define K1_X    131072u
#define K1_Y    139264u
#define K1_P    147456u
#define K1_V    155648u
#define K1_BP   163840u
#define K1_BV   164352u
#define K1_LDS  164864u

__global__ __launch_bounds__(NTHREADS) void k_proj(
    const float* __restrict__ price, const float* __restrict__ liquid,
    const float* __restrict__ W_p, const float* __restrict__ b_p,
    const float* __restrict__ W_v, const float* __restrict__ b_v,
    const float* __restrict__ W1,
    half_t* __restrict__ ws_p, half_t* __restrict__ ws_v,
    float* __restrict__ ws_PA, float* __restrict__ ws_VB) {
  extern __shared__ char smem[];
  half_t* sWp  = (half_t*)(smem + K1_WP);
  half_t* sWv  = (half_t*)(smem + K1_WV);
  half_t* sW1a = (half_t*)(smem + K1_W1A);
  half_t* sW1b = (half_t*)(smem + K1_W1B);
  half_t* sX   = (half_t*)(smem + K1_X);
  half_t* sY   = (half_t*)(smem + K1_Y);
  half_t* sP   = (half_t*)(smem + K1_P);
  half_t* sV   = (half_t*)(smem + K1_V);
  float*  sbp  = (float*)(smem + K1_BP);
  float*  sbv  = (float*)(smem + K1_BV);
  float*  sPAo = (float*)(smem + K1_WP);   // phase-2 staging (aliases sWp)
  float*  sVBo = (float*)(smem + K1_WV);   // phase-2 staging (aliases sWv)

  const int bt = blockIdx.x;
  const int tid = threadIdx.x, lane = tid & 31, w = tid >> 5;

  cvtCopyF16(price + (size_t)bt * 4096, sX, 4096, tid);
  cvtCopyF16(liquid + (size_t)bt * 4096, sY, 4096, tid);
  swizzleW(W_p, sWp, tid);
  swizzleW(W_v, sWv, tid);
  swizzleW(W1, sW1a, tid);               // rows   0..127 : p-part
  swizzleW(W1 + 128 * 128, sW1b, tid);   // rows 128..255 : v-part
  copyF32(b_p, sbp, 128, tid);
  copyF32(b_v, sbv, 128, tid);
  __syncthreads();

  const int hi8 = (lane & 16) ? 8 : 0;
  const int col = lane & 15;

  // Phase 1: p = X@W_p + b_p, v = Y@W_v + b_v   (32 output tiles, 4/wave)
  for (int t = w * 4; t < w * 4 + 4; ++t) {
    const int mat = t >> 4, rt = (t >> 3) & 1, c = t & 7;
    const half_t* Xs = mat ? sY : sX;
    const half_t* Ws = mat ? sWv : sWp;
    const float* bias = mat ? sbv : sbp;
    const int cc = c * 16 + col;
    v8f acc;
#pragma unroll
    for (int r = 0; r < 8; ++r) acc[r] = bias[cc];
    const half_t* arow = Xs + (rt * 16 + col) * 128;
#pragma unroll
    for (int kk = 0; kk < 4; ++kk)
      acc = wmma16(loadArow(arow, kk, lane), loadB(Ws, c, kk, lane), acc);
    half_t* Ps = mat ? sV : sP;
#pragma unroll
    for (int r = 0; r < 8; ++r)
      Ps[(rt * 16 + r + hi8) * 128 + cc] = (half_t)acc[r];
  }
  __syncthreads();

  // Stream p, v out asynchronously (contiguous b128) while phase 2 runs.
  asyncL2G(ws_p + (size_t)bt * 4096, K1_P, 8192, tid);
  asyncL2G(ws_v + (size_t)bt * 4096, K1_V, 8192, tid);

  // Phase 2: PA = p@W1a, VB = v@W1b  (bias b1 is added in kernel 2)
  for (int t = w * 4; t < w * 4 + 4; ++t) {
    const int mat = t >> 4, rt = (t >> 3) & 1, c = t & 7;
    const half_t* Xs = mat ? sV : sP;
    const half_t* Ws = mat ? sW1b : sW1a;
    const int cc = c * 16 + col;
    v8f acc = {};
    const half_t* arow = Xs + (rt * 16 + col) * 128;
#pragma unroll
    for (int kk = 0; kk < 4; ++kk)
      acc = wmma16(loadArow(arow, kk, lane), loadB(Ws, c, kk, lane), acc);
    float* stg = mat ? sVBo : sPAo;
#pragma unroll
    for (int r = 0; r < 8; ++r)
      stg[(rt * 16 + r + hi8) * 128 + cc] = acc[r];
  }
  __syncthreads();
  asyncL2G(ws_PA + (size_t)bt * 4096, K1_WP, 16384, tid);
  asyncL2G(ws_VB + (size_t)bt * 4096, K1_WV, 16384, tid);
  // S_ENDPGM performs an implicit wait-idle, draining ASYNCcnt.
}

// ---------------------------------------------------------------------------
// Kernel 2 LDS layout (byte offsets):
//   0       sW1c  32KB      114688  sPA 16KB      148480  sh    32KB
//   32768   sW1d  32KB      131072  sVB 16KB      181248  soutf 64KB
//   65536   sW2   32KB      147456  sb1 512B
//   98304   sp     8KB      147968  sb2 512B      total 245.. -> K2_LDS
//   106496  sv     8KB
// ---------------------------------------------------------------------------
#define K2_W1C  0u
#define K2_W1D  32768u
#define K2_W2   65536u
#define K2_P    98304u
#define K2_V    106496u
#define K2_PA   114688u
#define K2_VB   131072u
#define K2_B1   147456u
#define K2_B2   147968u
#define K2_H    148480u
#define K2_OUT  181248u
#define K2_LDS  246784u

__global__ __launch_bounds__(NTHREADS) void k_pair(
    const half_t* __restrict__ ws_p, const half_t* __restrict__ ws_v,
    const float* __restrict__ ws_PA, const float* __restrict__ ws_VB,
    const float* __restrict__ W1, const float* __restrict__ b1,
    const float* __restrict__ W2, const float* __restrict__ b2,
    float* __restrict__ out) {
  extern __shared__ char smem[];
  half_t* sW1c = (half_t*)(smem + K2_W1C);
  half_t* sW1d = (half_t*)(smem + K2_W1D);
  half_t* sW2  = (half_t*)(smem + K2_W2);
  half_t* sp   = (half_t*)(smem + K2_P);
  half_t* sv   = (half_t*)(smem + K2_V);
  float*  sPA  = (float*)(smem + K2_PA);
  float*  sVB  = (float*)(smem + K2_VB);
  float*  sb1  = (float*)(smem + K2_B1);
  float*  sb2  = (float*)(smem + K2_B2);
  half_t* sh   = (half_t*)(smem + K2_H);    // 8 waves x 16x128 f16
  float*  sout = (float*)(smem + K2_OUT);   // 8 waves x 16x128 f32

  const int blk = blockIdx.x;
  const int bt = blk >> 2, seg = blk & 3;
  const int tid = threadIdx.x, lane = tid & 31, w = tid >> 5;

  // Async-fill p/v/PA/VB straight into LDS (no VGPR round trip).
  asyncG2L(K2_P,  ws_p  + (size_t)bt * 4096, 8192, tid);
  asyncG2L(K2_V,  ws_v  + (size_t)bt * 4096, 8192, tid);
  asyncG2L(K2_PA, ws_PA + (size_t)bt * 4096, 16384, tid);
  asyncG2L(K2_VB, ws_VB + (size_t)bt * 4096, 16384, tid);
  // Weight swizzles need f32->f16 conversion, so they go through VGPRs.
  swizzleW(W1 + 256 * 128, sW1c, tid);   // rows 256..383 : prod
  swizzleW(W1 + 384 * 128, sW1d, tid);   // rows 384..511 : absdiff
  swizzleW(W2, sW2, tid);
  copyF32(b1, sb1, 128, tid);
  copyF32(b2, sb2, 128, tid);
  wait_async0();
  __syncthreads();

  const int hi8 = (lane & 16) ? 8 : 0;
  const int col = lane & 15;
  half_t* hbuf = sh + w * (16 * 128);        // wave-private h tile (f16)
  float*  obuf = sout + w * (16 * 128);      // wave-private out tile (f32)
  const uint32_t obuf_off = K2_OUT + (uint32_t)w * 8192u;

  for (int ti = 0; ti < 2; ++ti) {
    const int tile = seg * 16 + w * 2 + ti;   // 0..63 over this (b,t)
    const int n = tile >> 1;
    const int m0 = (tile & 1) * 16;
    const half_t* prow = sp + n * 128;
    const half_t* vrow = sv + (m0 + col) * 128;

    v16h aP[4], aD[4];
#pragma unroll
    for (int kk = 0; kk < 4; ++kk) buildPV(prow, vrow, kk, lane, aP[kk], aD[kk]);

    // Stage 1: h = silu(PA[n] + VB[m] + b1 + prod@W1c + absdiff@W1d)
#pragma unroll
    for (int c = 0; c < 8; ++c) {
      const int cc = c * 16 + col;
      v8f acc;
      const float pa = sPA[n * 128 + cc] + sb1[cc];
#pragma unroll
      for (int r = 0; r < 8; ++r)
        acc[r] = pa + sVB[(m0 + r + hi8) * 128 + cc];
#pragma unroll
      for (int kk = 0; kk < 4; ++kk) {
        acc = wmma16(aP[kk], loadB(sW1c, c, kk, lane), acc);
        acc = wmma16(aD[kk], loadB(sW1d, c, kk, lane), acc);
      }
#pragma unroll
      for (int r = 0; r < 8; ++r) {
        float x = acc[r];
        float s = x / (1.0f + __expf(-x));           // SiLU
        hbuf[(r + hi8) * 128 + cc] = (half_t)s;
      }
    }

    // Stage 2: out = h@W2 + b2   (re-fragment h from wave-private LDS)
    v16h aH[4];
#pragma unroll
    for (int kk = 0; kk < 4; ++kk)
      aH[kk] = loadArow(hbuf + col * 128, kk, lane);

    // Previous tile's async stores must finish reading obuf before reuse.
    wait_async0();
#pragma unroll
    for (int c2 = 0; c2 < 8; ++c2) {
      const int cc = c2 * 16 + col;
      v8f acc;
#pragma unroll
      for (int r = 0; r < 8; ++r) acc[r] = sb2[cc];
#pragma unroll
      for (int kk = 0; kk < 4; ++kk)
        acc = wmma16(aH[kk], loadB(sW2, c2, kk, lane), acc);
#pragma unroll
      for (int r = 0; r < 8; ++r)
        obuf[(r + hi8) * 128 + cc] = acc[r];
    }

    // Stream this wave's 16x128 f32 tile (8 KB, contiguous) to HBM.
    char* obase = (char*)(out + ((size_t)bt * 1024 + (size_t)n * 32 + m0) * 128);
    wait_ds0();   // async engine must observe our ds stores (separate pipe)
#pragma unroll
    for (int i = 0; i < 16; ++i)
      async_store_b128(obase + i * 512 + lane * 16,
                       obuf_off + (uint32_t)(i * 512 + lane * 16));
    // S_ENDPGM's implicit wait-idle drains the final tile's stores.
  }
}

// ---------------------------------------------------------------------------
extern "C" void kernel_launch(void* const* d_in, const int* in_sizes, int n_in,
                              void* d_out, int out_size, void* d_ws, size_t ws_size,
                              hipStream_t stream) {
  const float* price  = (const float*)d_in[0];
  const float* liquid = (const float*)d_in[1];
  const float* W_p    = (const float*)d_in[2];
  const float* b_p    = (const float*)d_in[3];
  const float* W_v    = (const float*)d_in[4];
  const float* b_v    = (const float*)d_in[5];
  const float* W1     = (const float*)d_in[6];
  const float* b1     = (const float*)d_in[7];
  const float* W2     = (const float*)d_in[8];
  const float* b2     = (const float*)d_in[9];
  float* out = (float*)d_out;

  char* ws = (char*)d_ws;                     // needs 12 MiB
  half_t* ws_p  = (half_t*)ws;                // 2 MiB  p  f16
  half_t* ws_v  = (half_t*)(ws + (2u << 20)); // 2 MiB  v  f16
  float*  ws_PA = (float*)(ws + (4u << 20));  // 4 MiB  p@W1a f32
  float*  ws_VB = (float*)(ws + (8u << 20));  // 4 MiB  v@W1b f32

  k_proj<<<256, NTHREADS, K1_LDS, stream>>>(price, liquid, W_p, b_p, W_v, b_v,
                                            W1, ws_p, ws_v, ws_PA, ws_VB);
  k_pair<<<1024, NTHREADS, K2_LDS, stream>>>(ws_p, ws_v, ws_PA, ws_VB,
                                             W1, b1, W2, b2, out);
}
